// MultiHeadAttention_13632226197515
// MI455X (gfx1250) — matrix operand order repaired and checked by reading, not compiled, not run on hardware
//
#include <hip/hip_runtime.h>

// ---------------------------------------------------------------------------
// MHA forward for MI455X (gfx1250, wave32, WMMA + async-to-LDS).
// Pipeline:
//   1) cvt_bf16: one-shot fp32->bf16 for activations + weights
//   2) proj_gemm<MODE>: LDS-free bf16 WMMA GEMM (Q,K,V proj + output proj)
//   3) flash_attn: online-softmax attention, K/V tiles double-buffered in LDS
//      via global_load_async_to_lds_b128 (ASYNCcnt) shared by all 4 waves
// ---------------------------------------------------------------------------

typedef __attribute__((ext_vector_type(16))) __bf16   v16bf;
typedef __attribute__((ext_vector_type(8)))  float    v8f;
typedef __attribute__((ext_vector_type(4)))  float    v4f;
typedef __attribute__((ext_vector_type(2)))  unsigned uint2v;
typedef __attribute__((ext_vector_type(4)))  unsigned uint4v;

union FragBF {
    v16bf          v;       // WMMA 16-bit A/B operand (8 VGPRs)
    unsigned short u[16];
    uint2v         d2[4];
    uint4v         d4[2];
};

__device__ __forceinline__ unsigned short f2bf(float f) {
    union { float f; unsigned u; } x;
    x.f = f;
    unsigned r = x.u + 0x7FFFu + ((x.u >> 16) & 1u);   // round-to-nearest-even
    return (unsigned short)(r >> 16);
}

constexpr int D_MODEL = 1024;
constexpr int NUM_HEADS = 16;
constexpr int D_K = 64;
constexpr int SEQ = 2048;
constexpr int BATCH = 2;
constexpr int GEMM_M = BATCH * SEQ;                 // 4096
constexpr size_t ACT_N = (size_t)GEMM_M * D_MODEL;  // 4M elems
constexpr size_t W_N   = (size_t)D_MODEL * D_MODEL; // 1M elems

// ---------------------------------------------------------------------------
// cvt_bf16: fp32 -> bf16, 8 elements/thread, exact grids (n % 2048 == 0)
// ---------------------------------------------------------------------------
__global__ __launch_bounds__(256)
void cvt_bf16(const float* __restrict__ src, unsigned short* __restrict__ dst)
{
    const size_t i = ((size_t)blockIdx.x * 256 + threadIdx.x) * 8;
    v4f a = *reinterpret_cast<const v4f*>(src + i);
    v4f b = *reinterpret_cast<const v4f*>(src + i + 4);
    union { unsigned short u[8]; uint4v q; } o;
#pragma unroll
    for (int j = 0; j < 4; ++j) { o.u[j] = f2bf(a[j]); o.u[4 + j] = f2bf(b[j]); }
    *reinterpret_cast<uint4v*>(dst + i) = o.q;
}

// ---------------------------------------------------------------------------
// proj_gemm: Y[m,n] = sum_k X[m,k] * W[n,k] + bias[n]   (bf16 in, f32 accum)
//   MODE 0: out bf16 [B,H,S,dk]      (Q, K projections)
//   MODE 1: out bf16 [B,H,dk,S]      (V projection, transposed for PV GEMM)
//   MODE 2: out f32  [M,N]           (final output projection)
// LDS-free: fragments loaded directly from global (weights L2-resident).
// Block 256 = 8 waves (4 M-waves x 2 N-waves); block tile 128M x 128N;
// wave tile 32M x 64N = 8 accum tiles; K-step 32 -> 8 WMMA / 12 b128 loads.
// ---------------------------------------------------------------------------
template <int MODE>
__global__ __launch_bounds__(256)
void proj_gemm(const unsigned short* __restrict__ X,
               const unsigned short* __restrict__ Wb,
               const float* __restrict__ bias, void* __restrict__ outp)
{
    constexpr int K = D_MODEL;
    const int lane = threadIdx.x & 31;
    const int wave = threadIdx.x >> 5;
    const int half = lane >> 4;
    const int l16  = lane & 15;
    const int wm   = wave & 3;
    const int wn   = wave >> 2;
    const int mbase = blockIdx.x * 128 + wm * 32;
    const int nbase = blockIdx.y * 128 + wn * 64;

    v8f acc[2][4];
#pragma unroll
    for (int r = 0; r < 2; ++r)
#pragma unroll
        for (int t = 0; t < 4; ++t) acc[r][t] = (v8f){};

    const unsigned short* xrow0 = X + (size_t)(mbase + l16) * K;
    const unsigned short* xrow1 = X + (size_t)(mbase + 16 + l16) * K;

    for (int kb = 0; kb < K; kb += 32) {
        // A fragments: lane=row, k = (j&8)<<1 | half<<3 | (j&7) -> two 16B runs
        FragBF a0, a1;
        {
            const uint4v* p0 = reinterpret_cast<const uint4v*>(xrow0 + kb);
            const uint4v* p1 = reinterpret_cast<const uint4v*>(xrow1 + kb);
            a0.d4[0] = p0[half];     a0.d4[1] = p0[2 + half];
            a1.d4[0] = p1[half];     a1.d4[1] = p1[2 + half];
        }
#pragma unroll
        for (int t = 0; t < 4; ++t) {
            // B fragment: lane N = l16, k = (j&8)<<1 | half<<3 | (j&7) -> two 16B runs
            FragBF b;
            const uint4v* brow = reinterpret_cast<const uint4v*>(
                Wb + (size_t)(nbase + t * 16 + l16) * K + kb);
            b.d4[0] = brow[half];
            b.d4[1] = brow[2 + half];
            acc[0][t] = __builtin_amdgcn_wmma_f32_16x16x32_bf16(
                false, a0.v, false, b.v, (short)0, acc[0][t], false, false);
            acc[1][t] = __builtin_amdgcn_wmma_f32_16x16x32_bf16(
                false, a1.v, false, b.v, (short)0, acc[1][t], false, false);
        }
    }

    // epilogue: C/D layout lane N = l16, VGPR i -> M = i + 8*half
#pragma unroll
    for (int t = 0; t < 4; ++t) {
        const int nl = nbase + t * 16 + l16;
        const float bv = bias[nl];
#pragma unroll
        for (int r = 0; r < 2; ++r) {
#pragma unroll
            for (int i = 0; i < 8; ++i) {
                const int ml = mbase + r * 16 + i + 8 * half;
                const float val = acc[r][t][i] + bv;
                if constexpr (MODE == 2) {
                    ((float*)outp)[(size_t)ml * D_MODEL + nl] = val;
                } else {
                    const int b = ml / SEQ, s = ml % SEQ;
                    const int h = nl >> 6, d = nl & 63;
                    size_t idx;
                    if constexpr (MODE == 0)
                        idx = (((size_t)(b * NUM_HEADS + h)) * SEQ + s) * D_K + d;
                    else
                        idx = (((size_t)(b * NUM_HEADS + h)) * D_K + d) * SEQ + s;
                    ((unsigned short*)outp)[idx] = f2bf(val);
                }
            }
        }
    }
}

// ---------------------------------------------------------------------------
// flash_attn: per (b,h), online-softmax attention.
//   Qb,Kb: bf16 [B,H,S,dk]; Vt: bf16 [B,H,dk,S]; X2: bf16 [B,S,H*dk]
// Block 128 = 4 waves, each owns a 16-query strip. K/V 64-key tiles are
// double-buffered in LDS via global_load_async_to_lds_b128 and shared by
// all 4 waves (4x cut in L2 traffic); copy of tile i+1 overlaps compute.
// ---------------------------------------------------------------------------
__global__ __launch_bounds__(128)
void flash_attn(const unsigned short* __restrict__ Qb,
                const unsigned short* __restrict__ Kb,
                const unsigned short* __restrict__ Vt,
                const int* __restrict__ mask,
                unsigned short* __restrict__ X2)
{
    __shared__ unsigned short Kt[2][64][64];   // [buf][key][dk]
    __shared__ unsigned short Vs[2][64][64];   // [buf][dk][key]
    __shared__ unsigned short Ps[4][16][64];   // wave-private P strips

    const int tid  = threadIdx.x;
    const int lane = tid & 31;
    const int wave = tid >> 5;
    const int half = lane >> 4;
    const int l16  = lane & 15;
    const int bh   = blockIdx.y;           // b*H + h
    const int b    = bh >> 4;
    const int h    = bh & 15;
    const int qbase = blockIdx.x * 64 + wave * 16;

    const unsigned short* qp = Qb + (size_t)bh * SEQ * D_K;
    const unsigned short* kp = Kb + (size_t)bh * SEQ * D_K;
    const unsigned short* vp = Vt + (size_t)bh * D_K * SEQ;
    const int* mp = mask + b * SEQ;

    // async stage of one 64x64 K tile + 64x64 V^T tile (512 x 16B chunks each,
    // 128 threads x 4 chunks); tracked with ASYNCcnt
    auto stage_tiles = [&](int buf, int kb) {
#pragma unroll
        for (int i = 0; i < 4; ++i) {
            const int c   = i * 128 + tid;       // chunk id 0..511
            const int row = c >> 3;              // 0..63
            const int col = (c & 7) * 8;         // halves
            unsigned lk = (unsigned)(unsigned long long)&Kt[buf][row][col];
            unsigned long long gk =
                (unsigned long long)(kp + (size_t)(kb + row) * D_K + col);
            asm volatile("global_load_async_to_lds_b128 %0, %1, off"
                         :: "v"(lk), "v"(gk) : "memory");
            unsigned lv = (unsigned)(unsigned long long)&Vs[buf][row][col];
            unsigned long long gv =
                (unsigned long long)(vp + (size_t)row * SEQ + kb + col);
            asm volatile("global_load_async_to_lds_b128 %0, %1, off"
                         :: "v"(lv), "v"(gv) : "memory");
        }
    };

    // preload Q A-fragments (dk=64 -> 2 K-steps) straight from global
    FragBF aq[2];
    {
        const uint4v* qrow = reinterpret_cast<const uint4v*>(qp + (size_t)(qbase + l16) * D_K);
#pragma unroll
        for (int ks = 0; ks < 2; ++ks) {
            aq[ks].d4[0] = qrow[ks * 4 + half];
            aq[ks].d4[1] = qrow[ks * 4 + 2 + half];
        }
    }

    float mrow[8], lrow[8];
    v8f o[4];
#pragma unroll
    for (int i = 0; i < 8; ++i) { mrow[i] = -1e30f; lrow[i] = 0.f; }
#pragma unroll
    for (int t = 0; t < 4; ++t) o[t] = (v8f){};

    const float scale = 0.125f;   // 1/sqrt(64)

    stage_tiles(0, 0);

    for (int kb = 0, it = 0; kb < SEQ; kb += 64, ++it) {
        const int buf = it & 1;
        asm volatile("s_wait_asynccnt 0" ::: "memory");
        __syncthreads();
        if (kb + 64 < SEQ) stage_tiles(buf ^ 1, kb + 64);

        // ---- logits S = Q K^T : 4 key sub-tiles x 2 K-steps (from LDS)
        v8f s[4];
#pragma unroll
        for (int t = 0; t < 4; ++t) {
            v8f c = (v8f){};
            const uint4v* krow = reinterpret_cast<const uint4v*>(&Kt[buf][t * 16 + l16][0]);
#pragma unroll
            for (int ks = 0; ks < 2; ++ks) {
                FragBF bk;                               // lane N=key, k table order
                bk.d4[0] = krow[ks * 4 + half];
                bk.d4[1] = krow[ks * 4 + 2 + half];
                c = __builtin_amdgcn_wmma_f32_16x16x32_bf16(
                    false, aq[ks].v, false, bk.v, (short)0, c, false, false);
            }
            s[t] = c;
        }

        // ---- key-padding mask, scale, online softmax
        float pm[4];
#pragma unroll
        for (int t = 0; t < 4; ++t)
            pm[t] = mp[kb + t * 16 + l16] ? 0.0f : -1.0e9f;

        float alpha[8];
#pragma unroll
        for (int i = 0; i < 8; ++i) {
            float mx = -1e30f;
#pragma unroll
            for (int t = 0; t < 4; ++t) {
                float v = s[t][i] * scale + pm[t];
                s[t][i] = v;
                mx = fmaxf(mx, v);
            }
            mx = fmaxf(mx, __shfl_xor(mx, 1, 32));
            mx = fmaxf(mx, __shfl_xor(mx, 2, 32));
            mx = fmaxf(mx, __shfl_xor(mx, 4, 32));
            mx = fmaxf(mx, __shfl_xor(mx, 8, 32));
            const float mnew = fmaxf(mrow[i], mx);
            alpha[i] = __expf(mrow[i] - mnew);
            mrow[i] = mnew;
            float rs = 0.f;
#pragma unroll
            for (int t = 0; t < 4; ++t) {
                float p = __expf(s[t][i] - mnew);
                s[t][i] = p;
                rs += p;
            }
            rs += __shfl_xor(rs, 1, 32);
            rs += __shfl_xor(rs, 2, 32);
            rs += __shfl_xor(rs, 4, 32);
            rs += __shfl_xor(rs, 8, 32);
            lrow[i] = lrow[i] * alpha[i] + rs;
        }
#pragma unroll
        for (int t = 0; t < 4; ++t)
#pragma unroll
            for (int i = 0; i < 8; ++i) o[t][i] *= alpha[i];

        // ---- C-layout P -> LDS -> A-layout fragments (wave-private, DS in-order)
#pragma unroll
        for (int t = 0; t < 4; ++t)
#pragma unroll
            for (int i = 0; i < 8; ++i)
                Ps[wave][i + 8 * half][t * 16 + l16] = f2bf(s[t][i]);

        FragBF ap[2];
        {
            const uint2v* prow = reinterpret_cast<const uint2v*>(&Ps[wave][l16][0]);
#pragma unroll
            for (int ks = 0; ks < 2; ++ks) {
                ap[ks].d2[0] = prow[ks * 8 + half * 2 + 0];
                ap[ks].d2[1] = prow[ks * 8 + half * 2 + 1];
                ap[ks].d2[2] = prow[ks * 8 + 4 + half * 2 + 0];
                ap[ks].d2[3] = prow[ks * 8 + 4 + half * 2 + 1];
            }
        }

        // ---- O += P * V (V^T tile from LDS: lane N=d, k=key table order)
#pragma unroll
        for (int dt = 0; dt < 4; ++dt) {
            const uint4v* vrow = reinterpret_cast<const uint4v*>(&Vs[buf][dt * 16 + l16][0]);
#pragma unroll
            for (int ks = 0; ks < 2; ++ks) {
                FragBF bv;
                bv.d4[0] = vrow[ks * 4 + half];
                bv.d4[1] = vrow[ks * 4 + 2 + half];
                o[dt] = __builtin_amdgcn_wmma_f32_16x16x32_bf16(
                    false, ap[ks].v, false, bv.v, (short)0, o[dt], false, false);
            }
        }
    }

    // ---- finalize: O /= l, store bf16 to [B,S,H*dk]
#pragma unroll
    for (int t = 0; t < 4; ++t) {
#pragma unroll
        for (int i = 0; i < 8; ++i) {
            const float val = o[t][i] / lrow[i];
            const int qr = qbase + i + 8 * half;
            const size_t idx = ((size_t)(b * SEQ + qr)) * D_MODEL + h * 64 + t * 16 + l16;
            X2[idx] = f2bf(val);
        }
    }
}

// ---------------------------------------------------------------------------
extern "C" void kernel_launch(void* const* d_in, const int* in_sizes, int n_in,
                              void* d_out, int out_size, void* d_ws, size_t ws_size,
                              hipStream_t stream)
{
    (void)in_sizes; (void)n_in; (void)out_size; (void)ws_size;

    const float* q     = (const float*)d_in[0];
    const float* k     = (const float*)d_in[1];
    const float* v     = (const float*)d_in[2];
    const int*   pmask = (const int*)  d_in[3];
    const float* wq    = (const float*)d_in[4];
    const float* bq    = (const float*)d_in[5];
    const float* wk    = (const float*)d_in[6];
    const float* bk    = (const float*)d_in[7];
    const float* wv    = (const float*)d_in[8];
    const float* bv    = (const float*)d_in[9];
    const float* wo    = (const float*)d_in[10];
    const float* bo    = (const float*)d_in[11];

    // workspace layout (bf16/ushort elements), total 32M elems = 64 MB
    unsigned short* ws = (unsigned short*)d_ws;
    unsigned short* qbf = ws;                    // 4M
    unsigned short* kbf = qbf + ACT_N;           // 4M
    unsigned short* vbf = kbf + ACT_N;           // 4M
    unsigned short* wqb = vbf + ACT_N;           // 1M
    unsigned short* wkb = wqb + W_N;             // 1M
    unsigned short* wvb = wkb + W_N;             // 1M
    unsigned short* wob = wvb + W_N;             // 1M
    unsigned short* Qb  = wob + W_N;             // 4M
    unsigned short* Kb  = Qb + ACT_N;            // 4M
    unsigned short* Vt  = Kb + ACT_N;            // 4M
    unsigned short* X2  = Vt + ACT_N;            // 4M
    float* out = (float*)d_out;

    dim3 blk256(256), blk128(128);

    // 1) one-shot bf16 conversions
    cvt_bf16<<<ACT_N / 2048, blk256, 0, stream>>>(q,  qbf);
    cvt_bf16<<<ACT_N / 2048, blk256, 0, stream>>>(k,  kbf);
    cvt_bf16<<<ACT_N / 2048, blk256, 0, stream>>>(v,  vbf);
    cvt_bf16<<<W_N   / 2048, blk256, 0, stream>>>(wq, wqb);
    cvt_bf16<<<W_N   / 2048, blk256, 0, stream>>>(wk, wkb);
    cvt_bf16<<<W_N   / 2048, blk256, 0, stream>>>(wv, wvb);
    cvt_bf16<<<W_N   / 2048, blk256, 0, stream>>>(wo, wob);

    // 2) projections
    dim3 gridG(GEMM_M / 128, D_MODEL / 128);     // 32 x 8
    proj_gemm<0><<<gridG, blk256, 0, stream>>>(qbf, wqb, bq, Qb);
    proj_gemm<0><<<gridG, blk256, 0, stream>>>(kbf, wkb, bk, Kb);
    proj_gemm<1><<<gridG, blk256, 0, stream>>>(vbf, wvb, bv, Vt);

    // 3) attention
    flash_attn<<<dim3(SEQ / 64, BATCH * NUM_HEADS), blk128, 0, stream>>>(
        Qb, Kb, Vt, pmask, X2);

    // 4) output projection (f32 result)
    proj_gemm<2><<<gridG, blk256, 0, stream>>>(X2, wob, bo, out);
}
